// DeformableDecoderCrossAttention_53747220742149
// MI455X (gfx1250) — matrix operand order repaired and checked by reading, not compiled, not run on hardware
//
#include <hip/hip_runtime.h>
#include <math.h>

#define BQ 8
#define NQ 900
#define CQ 256
#define HEADS 8
#define NPTS 4
#define HD 32
#define BEVH 200
#define BEVW 200
#define RADIUS 0.2f
#define NPROJ 96   // 64 offset cols + 32 weight cols

typedef __attribute__((ext_vector_type(16))) __bf16 v16bf;
typedef __attribute__((ext_vector_type(8)))  float  v8f;

// ---------------------------------------------------------------------------
// WMMA fragment loaders (contiguous K order k=16*(lane/16)+s in both A and B, wave32)
// ---------------------------------------------------------------------------

// A fragment: 16x32 bf16 tile from an fp32 row-major matrix X[ld], rows
// row0..row0+15 (clamped to maxrow for the N=900 tail), K window [k0,k0+32).
// Lane l<16: row=l, K = k0..k0+15; lane>=16: K = k0+16..k0+31.
__device__ inline v16bf load_a_f32(const float* __restrict__ X, int ld,
                                   int row0, int maxrow, int k0, int lane) {
    int r = row0 + (lane & 15);
    if (r > maxrow) r = maxrow;
    const float* p = X + (size_t)r * ld + k0 + ((lane & 16) ? 16 : 0);
    const float4* p4 = (const float4*)p;
    float4 f0 = p4[0], f1 = p4[1];   // K .. K+7
    float4 f2 = p4[2], f3 = p4[3];   // K+8 .. K+15
    v16bf a;
    a[0]=(__bf16)f0.x; a[1]=(__bf16)f0.y; a[2]=(__bf16)f0.z; a[3]=(__bf16)f0.w;
    a[4]=(__bf16)f1.x; a[5]=(__bf16)f1.y; a[6]=(__bf16)f1.z; a[7]=(__bf16)f1.w;
    a[8]=(__bf16)f2.x; a[9]=(__bf16)f2.y; a[10]=(__bf16)f2.z; a[11]=(__bf16)f2.w;
    a[12]=(__bf16)f3.x; a[13]=(__bf16)f3.y; a[14]=(__bf16)f3.z; a[15]=(__bf16)f3.w;
    return a;
}

// B fragment: 32x16 bf16 tile from a PRE-TRANSPOSED bf16 weight WT[n][k]
// (k stride CQ). Lane holds column n0+(lane&15); lanes 0-15 K=k0..k0+15,
// lanes 16-31 K=k0+16..k0+31. 16 consecutive bf16 = one aligned 32B load.
__device__ inline v16bf load_b_bf16T(const __bf16* __restrict__ WT,
                                     int n0, int k0, int lane) {
    const __bf16* p = WT + (size_t)(n0 + (lane & 15)) * CQ
                         + k0 + ((lane & 16) ? 16 : 0);
    return *(const v16bf*)p;   // base + offsets are 32B aligned by construction
}

// ---------------------------------------------------------------------------
// Kernel 0: transpose + convert weights to bf16 column-major in workspace
// ---------------------------------------------------------------------------
__global__ void prep_weights_kernel(const float* __restrict__ w_off,
                                    const float* __restrict__ w_wt,
                                    const float* __restrict__ w_out,
                                    __bf16* __restrict__ wprojT,
                                    __bf16* __restrict__ woutT) {
    int i = blockIdx.x * blockDim.x + threadIdx.x;
    if (i < NPROJ * CQ) {
        int j = i / CQ, k = i % CQ;
        float v = (j < 64) ? w_off[(size_t)k * 64 + j]
                           : w_wt[(size_t)k * 32 + (j - 64)];
        wprojT[(size_t)j * CQ + k] = (__bf16)v;
    }
    int i2 = i - NPROJ * CQ;
    if (i2 >= 0 && i2 < CQ * CQ) {
        int n = i2 / CQ, k = i2 % CQ;
        woutT[(size_t)n * CQ + k] = (__bf16)w_out[(size_t)k * CQ + n];
    }
}

// ---------------------------------------------------------------------------
// Kernel 1: proj = query @ [w_off|w_wt] + bias via WMMA. 3 waves per block,
// each wave owns TWO 16-col tiles (2 accumulators) so the A fragment (query)
// is loaded once and reused for 2 WMMAs per K-step. Epilogue computes
// loc = ref + tanh(off)*RADIUS ([B,N,64]) and raw weight logits ([B,N,32]).
// ---------------------------------------------------------------------------
__global__ void proj_kernel(const float* __restrict__ query,
                            const float* __restrict__ refpts,
                            const float* __restrict__ b_off,
                            const float* __restrict__ b_wt,
                            const __bf16* __restrict__ wprojT,
                            float* __restrict__ loc,
                            float* __restrict__ wlog) {
    const int mt   = blockIdx.x;           // 0..56 (ceil(900/16))
    const int b    = blockIdx.y;           // 0..7
    const int wave = threadIdx.x >> 5;     // 0..2
    const int lane = threadIdx.x & 31;
    const int n0   = wave * 32;            // two adjacent 16-col tiles
    const int row0 = mt * 16;
    const float* Q = query + (size_t)b * NQ * CQ;

    v8f c0 = {}, c1 = {};
    for (int k0 = 0; k0 < CQ; k0 += 32) {
        v16bf a  = load_a_f32(Q, CQ, row0, NQ - 1, k0, lane);
        v16bf b0 = load_b_bf16T(wprojT, n0,      k0, lane);
        v16bf b1 = load_b_bf16T(wprojT, n0 + 16, k0, lane);
        c0 = __builtin_amdgcn_wmma_f32_16x16x32_bf16(false, a, false, b0,
                                                     (short)0, c0, false, false);
        c1 = __builtin_amdgcn_wmma_f32_16x16x32_bf16(false, a, false, b1,
                                                     (short)0, c1, false, false);
    }

    const int mo = (lane & 16) ? 8 : 0;
#pragma unroll
    for (int j = 0; j < 2; ++j) {
        const v8f   c   = j ? c1 : c0;
        const int   n   = n0 + j * 16 + (lane & 15);
        const float bia = (n < 64) ? b_off[n] : b_wt[n - 64];
#pragma unroll
        for (int r = 0; r < 8; ++r) {
            int q = row0 + r + mo;
            if (q >= NQ) continue;
            float v = c[r] + bia;
            size_t bq = (size_t)b * NQ + q;
            if (n < 64) {
                // col j: head=j/8, p=(j/2)%4, coord=j&1 -> flat [B,N,64]
                float refv = refpts[bq * 2 + (n & 1)];
                loc[bq * 64 + n] = refv + tanhf(v) * RADIUS;
            } else {
                wlog[bq * 32 + (n - 64)] = v;
            }
        }
    }
}

// ---------------------------------------------------------------------------
// Kernel 2: softmax over P + bilinear gather. One wave per (b,q,head);
// lane = channel (hd=32), so each corner fetch is a coalesced 128B line.
// ---------------------------------------------------------------------------
__device__ inline float fetch_corner(const float* __restrict__ base,
                                     int xi, int yi) {
    if (xi < 0 || xi >= BEVW || yi < 0 || yi >= BEVH) return 0.f;
    return base[(size_t)(yi * BEVW + xi) * CQ];
}

__global__ void sample_kernel(const float* __restrict__ memory,
                              const float* __restrict__ loc,
                              const float* __restrict__ wlog,
                              float* __restrict__ fused) {
    const int gw   = blockIdx.x * (blockDim.x >> 5) + (threadIdx.x >> 5);
    const int lane = threadIdx.x & 31;
    if (gw >= BQ * NQ * HEADS) return;
    const int h = gw % HEADS;
    const int t = gw / HEADS;
    const int q = t % NQ;
    const int b = t / NQ;
    const size_t bq = (size_t)b * NQ + q;

    // softmax over the 4 points (uniform across lanes)
    float l0 = wlog[bq * 32 + h * 4 + 0];
    float l1 = wlog[bq * 32 + h * 4 + 1];
    float l2 = wlog[bq * 32 + h * 4 + 2];
    float l3 = wlog[bq * 32 + h * 4 + 3];
    float mx = fmaxf(fmaxf(l0, l1), fmaxf(l2, l3));
    float e0 = __expf(l0 - mx), e1 = __expf(l1 - mx);
    float e2 = __expf(l2 - mx), e3 = __expf(l3 - mx);
    float inv = 1.f / (e0 + e1 + e2 + e3);
    float w[4] = { e0 * inv, e1 * inv, e2 * inv, e3 * inv };

    const float* membase = memory + (size_t)b * (BEVH * BEVW) * CQ
                                  + h * HD + lane;
    float acc = 0.f;
#pragma unroll
    for (int p = 0; p < NPTS; ++p) {
        float x = loc[bq * 64 + h * 8 + p * 2 + 0] * BEVW - 0.5f;
        float y = loc[bq * 64 + h * 8 + p * 2 + 1] * BEVH - 0.5f;
        float x0f = floorf(x), y0f = floorf(y);
        int   x0 = (int)x0f, y0 = (int)y0f;
        float fx = x - x0f, fy = y - y0f;
        float v00 = fetch_corner(membase, x0,     y0);
        float v01 = fetch_corner(membase, x0 + 1, y0);
        float v10 = fetch_corner(membase, x0,     y0 + 1);
        float v11 = fetch_corner(membase, x0 + 1, y0 + 1);
        float bil = v00 * (1.f - fx) * (1.f - fy) + v01 * fx * (1.f - fy)
                  + v10 * (1.f - fx) * fy         + v11 * fx * fy;
        acc += bil * w[p];
    }
    fused[bq * CQ + h * HD + lane] = acc;
}

// ---------------------------------------------------------------------------
// Kernel 3: out = fused[7200,256] @ w_out[256,256] + b_out via WMMA.
// 7200 = 450 * 16 exactly. Block = 4 waves; each wave owns a 16x64 output
// strip (4 accumulators) so each A fragment feeds 4 WMMAs, and each block
// covers the full 256 output columns -> fused rows are read once per block.
// ---------------------------------------------------------------------------
__global__ void out_gemm_kernel(const float* __restrict__ fused,
                                const __bf16* __restrict__ woutT,
                                const float* __restrict__ b_out,
                                float* __restrict__ out) {
    const int mt   = blockIdx.x;            // 0..449
    const int wave = threadIdx.x >> 5;      // 0..3
    const int lane = threadIdx.x & 31;
    const int n0   = wave * 64;             // 16x64 strip per wave
    const int row0 = mt * 16;

    v8f c0 = {}, c1 = {}, c2 = {}, c3 = {};
    for (int k0 = 0; k0 < CQ; k0 += 32) {
        v16bf a  = load_a_f32(fused, CQ, row0, BQ * NQ - 1, k0, lane);
        v16bf b0 = load_b_bf16T(woutT, n0,      k0, lane);
        v16bf b1 = load_b_bf16T(woutT, n0 + 16, k0, lane);
        v16bf b2 = load_b_bf16T(woutT, n0 + 32, k0, lane);
        v16bf b3 = load_b_bf16T(woutT, n0 + 48, k0, lane);
        c0 = __builtin_amdgcn_wmma_f32_16x16x32_bf16(false, a, false, b0,
                                                     (short)0, c0, false, false);
        c1 = __builtin_amdgcn_wmma_f32_16x16x32_bf16(false, a, false, b1,
                                                     (short)0, c1, false, false);
        c2 = __builtin_amdgcn_wmma_f32_16x16x32_bf16(false, a, false, b2,
                                                     (short)0, c2, false, false);
        c3 = __builtin_amdgcn_wmma_f32_16x16x32_bf16(false, a, false, b3,
                                                     (short)0, c3, false, false);
    }

    const int mo = (lane & 16) ? 8 : 0;
#pragma unroll
    for (int j = 0; j < 4; ++j) {
        const v8f c = (j == 0) ? c0 : (j == 1) ? c1 : (j == 2) ? c2 : c3;
        const int n = n0 + j * 16 + (lane & 15);
        const float z = b_out[n];
#pragma unroll
        for (int r = 0; r < 8; ++r) {
            int m = row0 + r + mo;
            out[(size_t)m * CQ + n] = c[r] + z;
        }
    }
}

// ---------------------------------------------------------------------------
// Launch
// ---------------------------------------------------------------------------
extern "C" void kernel_launch(void* const* d_in, const int* in_sizes, int n_in,
                              void* d_out, int out_size, void* d_ws, size_t ws_size,
                              hipStream_t stream) {
    const float* query  = (const float*)d_in[0];
    const float* memory = (const float*)d_in[1];
    const float* refpts = (const float*)d_in[2];
    const float* w_off  = (const float*)d_in[3];
    const float* b_off  = (const float*)d_in[4];
    const float* w_wt   = (const float*)d_in[5];
    const float* b_wt   = (const float*)d_in[6];
    const float* w_out  = (const float*)d_in[7];
    const float* b_out  = (const float*)d_in[8];
    float* out = (float*)d_out;

    // workspace carve-up
    char* ws = (char*)d_ws;
    float* loc   = (float*)ws;                        // B*N*64 f32
    float* wlog  = loc  + (size_t)BQ * NQ * 64;       // B*N*32 f32
    float* fused = wlog + (size_t)BQ * NQ * 32;       // B*N*256 f32
    __bf16* wprojT = (__bf16*)(fused + (size_t)BQ * NQ * CQ);  // 96*256 bf16
    __bf16* woutT  = wprojT + (size_t)NPROJ * CQ;              // 256*256 bf16

    // 0) weight transpose+convert: 96*256 + 256*256 = 90112 elements
    {
        int total = NPROJ * CQ + CQ * CQ;
        prep_weights_kernel<<<(total + 255) / 256, 256, 0, stream>>>(
            w_off, w_wt, w_out, wprojT, woutT);
    }
    // 1) projection GEMM + tanh/ref/logit epilogue (3 waves x 2 tiles)
    {
        dim3 grid((NQ + 15) / 16, BQ);
        proj_kernel<<<grid, 96, 0, stream>>>(query, refpts, b_off, b_wt,
                                             wprojT, loc, wlog);
    }
    // 2) softmax + bilinear gather: 57600 waves, 8 waves/block
    {
        int waves = BQ * NQ * HEADS;
        sample_kernel<<<(waves + 7) / 8, 256, 0, stream>>>(memory, loc, wlog,
                                                           fused);
    }
    // 3) output GEMM + bias (4 waves x 16x64 strips)
    {
        out_gemm_kernel<<<(BQ * NQ) / 16, 128, 0, stream>>>(fused, woutT,
                                                            b_out, out);
    }
}
